// Block_8684423872838
// MI455X (gfx1250) — matrix operand order repaired and checked by reading, not compiled, not run on hardware
//
#include <hip/hip_runtime.h>

// ---------------------------------------------------------------------------
// CDNA5 (gfx1250) transformer block: LN1 -> QKV -> flash-attn -> proj+res ->
// LN2 -> MLP(fast_gelu)+res.  All GEMMs use v_wmma_f32_16x16x32_f16.
// GEMMs are LDS double-buffered: 1 barrier / 32-deep K tile, global loads for
// tile k+1 issued ahead of tile k's WMMAs, prefetch at k+2.
// ---------------------------------------------------------------------------

typedef __attribute__((ext_vector_type(16))) _Float16 v16h;
typedef __attribute__((ext_vector_type(8)))  _Float16 v8h;
typedef __attribute__((ext_vector_type(4)))  _Float16 v4h;
typedef __attribute__((ext_vector_type(8)))  float    v8f;

__device__ __forceinline__ v16h combine16(const _Float16* p0, const _Float16* p1) {
  v8h a = *(const v8h*)p0;
  v8h b = *(const v8h*)p1;
  v16h r;
#pragma unroll
  for (int i = 0; i < 8; ++i) { r[i] = a[i]; r[i + 8] = b[i]; }
  return r;
}

// ------------------------------ utility kernels -----------------------------

__global__ __launch_bounds__(256) void zero_kernel(uint4* __restrict__ p, long n16) {
  long i = (long)blockIdx.x * blockDim.x + threadIdx.x;
  const long stride = (long)gridDim.x * blockDim.x;
  uint4 z; z.x = 0u; z.y = 0u; z.z = 0u; z.w = 0u;
  for (; i < n16; i += stride) p[i] = z;
}

__global__ __launch_bounds__(256) void cvt_f32_to_f16(const float* __restrict__ w,
                                                      _Float16* __restrict__ o, int n4) {
  int i = blockIdx.x * blockDim.x + threadIdx.x;
  const int stride = gridDim.x * blockDim.x;
  for (; i < n4; i += stride) {
    float4 f = ((const float4*)w)[i];
    v4h h;
    h[0] = (_Float16)f.x; h[1] = (_Float16)f.y;
    h[2] = (_Float16)f.z; h[3] = (_Float16)f.w;
    ((v4h*)o)[i] = h;
  }
}

// LayerNorm over C=1024, one block (8 waves) per row, f16 output.
__global__ __launch_bounds__(256) void ln_kernel(const float* __restrict__ x,
                                                 const float* __restrict__ g,
                                                 const float* __restrict__ b,
                                                 _Float16* __restrict__ out) {
  const int C = 1024;
  const int m = blockIdx.x;
  const int tid = threadIdx.x;
  const float4 xv = ((const float4*)(x + (size_t)m * C))[tid];
  float s1 = xv.x + xv.y + xv.z + xv.w;
  float s2 = xv.x * xv.x + xv.y * xv.y + xv.z * xv.z + xv.w * xv.w;
#pragma unroll
  for (int off = 1; off < 32; off <<= 1) {
    s1 += __shfl_xor(s1, off, 32);
    s2 += __shfl_xor(s2, off, 32);
  }
  __shared__ float r1[8], r2[8];
  if ((tid & 31) == 0) { r1[tid >> 5] = s1; r2[tid >> 5] = s2; }
  __syncthreads();
  float t1 = 0.f, t2 = 0.f;
#pragma unroll
  for (int i = 0; i < 8; ++i) { t1 += r1[i]; t2 += r2[i]; }
  const float mu = t1 * (1.0f / C);
  const float var = t2 * (1.0f / C) - mu * mu;
  const float rs = rsqrtf(var + 1e-5f);
  const float4 gv = ((const float4*)g)[tid];
  const float4 bv = ((const float4*)b)[tid];
  v4h o4;
  o4[0] = (_Float16)((xv.x - mu) * rs * gv.x + bv.x);
  o4[1] = (_Float16)((xv.y - mu) * rs * gv.y + bv.y);
  o4[2] = (_Float16)((xv.z - mu) * rs * gv.z + bv.z);
  o4[3] = (_Float16)((xv.w - mu) * rs * gv.w + bv.w);
  ((v4h*)out)[(size_t)m * 256 + tid] = o4;
}

// ------------------------------ tiled WMMA GEMM -----------------------------
// C[M,N] = A[M,K](f16) * Bw[N,K](f16)^T, epilogue selected by EP:
//   0=Q (scaled, scatter [BH,608,64])  1=K (scatter)  2=V (scatter transposed)
//   3=proj (+bias +resid, f32 out)     4=fc1 (+bias, fast_gelu, f16 out)
//   5=fc2 (+bias +resid, f32 out)

template <int EP>
__global__ __launch_bounds__(256) void gemm_kernel(const _Float16* __restrict__ A,
                                                   const _Float16* __restrict__ Bw,
                                                   const float* __restrict__ bias,
                                                   const float* __restrict__ resid,
                                                   void* __restrict__ outp,
                                                   int M, int N, int K) {
  constexpr int BM = 128, BN = 128, BK = 32, LD = 40;
  __shared__ __align__(32) _Float16 As[2][BM * LD];
  __shared__ __align__(32) _Float16 Bs[2][BN * LD];
  const int tid = threadIdx.x;
  const int lane = tid & 31, wid = tid >> 5;
  const int wm = wid & 1, wn = wid >> 1;      // 2x4 wave grid -> 64x32 per wave
  const int hl = lane >> 4, l16 = lane & 15;  // half-of-wave, lane-in-half
  const int bm = blockIdx.x * BM, bn = blockIdx.y * BN;
  const int sRow = tid >> 2, sCol = (tid & 3) * 8;
  const int nk = K / BK;

  v8f acc[4][2];
#pragma unroll
  for (int i = 0; i < 4; ++i)
#pragma unroll
    for (int j = 0; j < 2; ++j)
#pragma unroll
      for (int r = 0; r < 8; ++r) acc[i][j][r] = 0.0f;

  // ---- preamble: stage K-tile 0 into buffer 0 ----
  uint4 na[2], nb[2];
#pragma unroll
  for (int p = 0; p < 2; ++p) {
    const int row = sRow + p * 64;
    const int gm = bm + row;
    uint4 av; av.x = 0u; av.y = 0u; av.z = 0u; av.w = 0u;
    if (gm < M) av = *(const uint4*)(A + (size_t)gm * K + sCol);
    na[p] = av;
    nb[p] = *(const uint4*)(Bw + (size_t)(bn + row) * K + sCol);
  }
#pragma unroll
  for (int p = 0; p < 2; ++p) {
    const int row = sRow + p * 64;
    *(uint4*)(&As[0][row * LD + sCol]) = na[p];
    *(uint4*)(&Bs[0][row * LD + sCol]) = nb[p];
  }

  for (int kt = 0; kt < nk; ++kt) {
    __syncthreads();
    const int buf = kt & 1;
    const bool more = (kt + 1) < nk;

    // ---- issue global loads for K-tile kt+1 early (latency hiding) ----
    if (more) {
      const int k1 = (kt + 1) * BK;
#pragma unroll
      for (int p = 0; p < 2; ++p) {
        const int row = sRow + p * 64;
        const int gm = bm + row;
        uint4 av; av.x = 0u; av.y = 0u; av.z = 0u; av.w = 0u;
        if (gm < M) av = *(const uint4*)(A + (size_t)gm * K + k1 + sCol);
        na[p] = av;
        nb[p] = *(const uint4*)(Bw + (size_t)(bn + row) * K + k1 + sCol);
      }
      if (kt + 2 < nk) {  // prefetch K-tile kt+2 (gfx1250 global_prefetch_b8)
        const int k2 = (kt + 2) * BK;
        if (bm + sRow < M)
          __builtin_prefetch(A + (size_t)(bm + sRow) * K + k2 + sCol, 0, 1);
        __builtin_prefetch(Bw + (size_t)(bn + sRow) * K + k2 + sCol, 0, 1);
      }
    }

    // ---- fragments per the CDNA5 WMMA lane layouts ----
    v16h af[4], bf[2];
#pragma unroll
    for (int i = 0; i < 4; ++i) {
      // A 16x32: lane<16 holds K 0..7 / 16..23; lane>=16 holds K 8..15 / 24..31
      const _Float16* p = &As[buf][(wm * 64 + i * 16 + l16) * LD + hl * 8];
      af[i] = combine16(p, p + 16);
    }
#pragma unroll
    for (int j = 0; j < 2; ++j) {
      // B 32x16: lane<16 = column, K 0..7 / 16..23; lane>=16 = column, K 8..15 / 24..31
      const _Float16* p = &Bs[buf][(wn * 32 + j * 16 + l16) * LD + hl * 8];
      bf[j] = combine16(p, p + 16);
    }
#pragma unroll
    for (int i = 0; i < 4; ++i)
#pragma unroll
      for (int j = 0; j < 2; ++j)
        acc[i][j] = __builtin_amdgcn_wmma_f32_16x16x32_f16(
            false, af[i], false, bf[j], (short)0, acc[i][j], false, false);

    // ---- commit K-tile kt+1 to the other buffer ----
    if (more) {
      const int nbuf = buf ^ 1;
#pragma unroll
      for (int p = 0; p < 2; ++p) {
        const int row = sRow + p * 64;
        *(uint4*)(&As[nbuf][row * LD + sCol]) = na[p];
        *(uint4*)(&Bs[nbuf][row * LD + sCol]) = nb[p];
      }
    }
  }

  // ------------------------------- epilogue -------------------------------
  const int colBase = bn + wn * 32 + l16;
  float bvj[2] = {0.0f, 0.0f};
  if constexpr (EP >= 3) {
    bvj[0] = bias[colBase];
    bvj[1] = bias[colBase + 16];
  }
#pragma unroll
  for (int i = 0; i < 4; ++i) {
#pragma unroll
    for (int r = 0; r < 8; ++r) {
      const int row = bm + wm * 64 + i * 16 + hl * 8 + r;  // D layout: M=r+8*hl
      if (row < M) {
        if constexpr (EP <= 2) {
          _Float16* o = (_Float16*)outp;
          const int b = row / 577, t = row - b * 577;  // hoisted per row
#pragma unroll
          for (int j = 0; j < 2; ++j) {
            const int col = colBase + j * 16;
            const int hh = col >> 6, d = col & 63;
            const float v = acc[i][j][r];
            if constexpr (EP == 0)
              o[((size_t)(b * 16 + hh) * 608 + t) * 64 + d] = (_Float16)(v * 0.125f);
            if constexpr (EP == 1)
              o[((size_t)(b * 16 + hh) * 608 + t) * 64 + d] = (_Float16)v;
            if constexpr (EP == 2)
              o[((size_t)(b * 16 + hh) * 64 + d) * 608 + t] = (_Float16)v;
          }
        } else {
#pragma unroll
          for (int j = 0; j < 2; ++j) {
            const int col = colBase + j * 16;
            const float v = acc[i][j][r];
            if constexpr (EP == 3 || EP == 5) {
              ((float*)outp)[(size_t)row * N + col] =
                  v + bvj[j] + resid[(size_t)row * N + col];
            } else {  // EP == 4: fast_gelu(x) = x * sigmoid(1.702 x)
              const float s = v + bvj[j];
              const float gl = s / (1.0f + __expf(-1.702f * s));
              ((_Float16*)outp)[(size_t)row * N + col] = (_Float16)gl;
            }
          }
        }
      }
    }
  }
}

// --------------------------- fused flash attention --------------------------
// grid(BH=256, 5); 8 waves/block, one 16-query tile per wave; key strips of 32.
__global__ __launch_bounds__(256) void attn_kernel(const _Float16* __restrict__ q,
                                                   const _Float16* __restrict__ k,
                                                   const _Float16* __restrict__ vt,
                                                   _Float16* __restrict__ o) {
  constexpr int NPAD = 608, D = 64;
  __shared__ __align__(32) _Float16 Pbuf[8][16 * 40];  // per-wave P restage
  const int tid = threadIdx.x, lane = tid & 31, wid = tid >> 5;
  const int hl = lane >> 4, l16 = lane & 15;
  const int bh = blockIdx.x;
  int qt = blockIdx.y * 8 + wid;
  if (qt > 36) qt = 36;  // duplicate-compute clamp keeps __syncthreads uniform
  const _Float16* qb = q + (size_t)bh * NPAD * D;
  const _Float16* kb = k + (size_t)bh * NPAD * D;
  const _Float16* vb = vt + (size_t)bh * D * NPAD;

  v16h qf[2];
#pragma unroll
  for (int ks = 0; ks < 2; ++ks) {
    const _Float16* p = qb + (size_t)(qt * 16 + l16) * D + ks * 32 + hl * 8;
    qf[ks] = combine16(p, p + 16);
  }

  v8f O[4];
#pragma unroll
  for (int dt = 0; dt < 4; ++dt)
#pragma unroll
    for (int r = 0; r < 8; ++r) O[dt][r] = 0.0f;
  float mrow[8], lrow[8];
#pragma unroll
  for (int r = 0; r < 8; ++r) { mrow[r] = -1e30f; lrow[r] = 0.0f; }

  for (int kt = 0; kt < 19; ++kt) {
    v8f S0, S1;
#pragma unroll
    for (int r = 0; r < 8; ++r) { S0[r] = 0.0f; S1[r] = 0.0f; }
#pragma unroll
    for (int ks = 0; ks < 2; ++ks) {  // contraction over D=64 in two x32 steps
      const _Float16* p0 = kb + (size_t)(kt * 32 + l16) * D + ks * 32 + hl * 8;
      const v16h kf0 = combine16(p0, p0 + 16);
      const _Float16* p1 = kb + (size_t)(kt * 32 + 16 + l16) * D + ks * 32 + hl * 8;
      const v16h kf1 = combine16(p1, p1 + 16);
      S0 = __builtin_amdgcn_wmma_f32_16x16x32_f16(false, qf[ks], false, kf0,
                                                  (short)0, S0, false, false);
      S1 = __builtin_amdgcn_wmma_f32_16x16x32_f16(false, qf[ks], false, kf1,
                                                  (short)0, S1, false, false);
    }
    const int c0 = kt * 32 + l16, c1 = c0 + 16;
    float alpha[8];
#pragma unroll
    for (int r = 0; r < 8; ++r) {
      const float s0 = (c0 < 577) ? S0[r] : -1e30f;
      const float s1 = (c1 < 577) ? S1[r] : -1e30f;
      float mt = fmaxf(s0, s1);
#pragma unroll
      for (int off = 1; off < 16; off <<= 1) mt = fmaxf(mt, __shfl_xor(mt, off, 32));
      const float mn = fmaxf(mrow[r], mt);
      const float a = __expf(mrow[r] - mn);
      const float p0 = __expf(s0 - mn);
      const float p1 = __expf(s1 - mn);
      float ps = p0 + p1;
#pragma unroll
      for (int off = 1; off < 16; off <<= 1) ps += __shfl_xor(ps, off, 32);
      lrow[r] = lrow[r] * a + ps;
      mrow[r] = mn;
      alpha[r] = a;
      Pbuf[wid][(hl * 8 + r) * 40 + l16] = (_Float16)p0;        // D-layout store
      Pbuf[wid][(hl * 8 + r) * 40 + 16 + l16] = (_Float16)p1;
    }
#pragma unroll
    for (int dt = 0; dt < 4; ++dt)
#pragma unroll
      for (int r = 0; r < 8; ++r) O[dt][r] *= alpha[r];
    __syncthreads();  // Pbuf cross-lane visibility (same-wave WAR is in-order)
    const _Float16* pp = &Pbuf[wid][l16 * 40 + hl * 8];          // A-frag read
    const v16h pf = combine16(pp, pp + 16);
#pragma unroll
    for (int dt = 0; dt < 4; ++dt) {
      const _Float16* vp = vb + (size_t)(dt * 16 + l16) * NPAD + kt * 32 + hl * 8;
      const v16h vf = combine16(vp, vp + 16);
      O[dt] = __builtin_amdgcn_wmma_f32_16x16x32_f16(false, pf, false, vf,
                                                     (short)0, O[dt], false, false);
    }
  }

  const int b = bh >> 4, hh = bh & 15;
#pragma unroll
  for (int r = 0; r < 8; ++r) {
    const int t = qt * 16 + hl * 8 + r;
    if (t < 577) {
      const float inv = 1.0f / lrow[r];
#pragma unroll
      for (int dt = 0; dt < 4; ++dt)
        o[(size_t)(b * 577 + t) * 1024 + hh * 64 + dt * 16 + l16] =
            (_Float16)(O[dt][r] * inv);
    }
  }
}

// --------------------------------- launcher ---------------------------------

extern "C" void kernel_launch(void* const* d_in, const int* in_sizes, int n_in,
                              void* d_out, int out_size, void* d_ws, size_t ws_size,
                              hipStream_t stream) {
  (void)in_sizes; (void)n_in; (void)out_size; (void)ws_size;
  const float* x   = (const float*)d_in[0];
  const float* g1  = (const float*)d_in[1];
  const float* be1 = (const float*)d_in[2];
  const float* Wq  = (const float*)d_in[3];
  const float* Wk  = (const float*)d_in[4];
  const float* Wv  = (const float*)d_in[5];
  const float* Wp  = (const float*)d_in[6];
  const float* bp  = (const float*)d_in[7];
  const float* g2  = (const float*)d_in[8];
  const float* be2 = (const float*)d_in[9];
  const float* W1  = (const float*)d_in[10];
  const float* b1  = (const float*)d_in[11];
  const float* W2  = (const float*)d_in[12];
  const float* b2  = (const float*)d_in[13];
  float* out = (float*)d_out;

  constexpr int M = 9232, C = 1024, H = 4096, BH = 256, NPAD = 608;
  char* ws = (char*)d_ws;
  size_t off = 0;
  auto alloc = [&](size_t bytes) {
    size_t o = off;
    off += (bytes + 255) & ~(size_t)255;
    return o;
  };
  const size_t o_h16 = alloc((size_t)M * C * 2);
  const size_t o_q   = alloc((size_t)BH * NPAD * 64 * 2);
  const size_t o_k   = alloc((size_t)BH * NPAD * 64 * 2);
  const size_t o_v   = alloc((size_t)BH * NPAD * 64 * 2);
  const size_t o_o   = alloc((size_t)M * C * 2);
  const size_t o_a16 = o_q;  // overlay: q/k/v/o are dead once MLP starts
  const size_t o_x1  = alloc((size_t)M * C * 4);
  const size_t o_wq  = alloc((size_t)C * C * 2);
  const size_t o_wk  = alloc((size_t)C * C * 2);
  const size_t o_wv  = alloc((size_t)C * C * 2);
  const size_t o_wp  = alloc((size_t)C * C * 2);
  const size_t o_w1  = alloc((size_t)H * C * 2);
  const size_t o_w2  = alloc((size_t)C * H * 2);

  _Float16* h16  = (_Float16*)(ws + o_h16);
  _Float16* q16  = (_Float16*)(ws + o_q);
  _Float16* k16  = (_Float16*)(ws + o_k);
  _Float16* v16  = (_Float16*)(ws + o_v);
  _Float16* o16  = (_Float16*)(ws + o_o);
  _Float16* a16  = (_Float16*)(ws + o_a16);
  float*    x1   = (float*)(ws + o_x1);
  _Float16* wq16 = (_Float16*)(ws + o_wq);
  _Float16* wk16 = (_Float16*)(ws + o_wk);
  _Float16* wv16 = (_Float16*)(ws + o_wv);
  _Float16* wp16 = (_Float16*)(ws + o_wp);
  _Float16* w116 = (_Float16*)(ws + o_w1);
  _Float16* w216 = (_Float16*)(ws + o_w2);

  // weight conversion f32 -> f16
  cvt_f32_to_f16<<<1024, 256, 0, stream>>>(Wq, wq16, C * C / 4);
  cvt_f32_to_f16<<<1024, 256, 0, stream>>>(Wk, wk16, C * C / 4);
  cvt_f32_to_f16<<<1024, 256, 0, stream>>>(Wv, wv16, C * C / 4);
  cvt_f32_to_f16<<<1024, 256, 0, stream>>>(Wp, wp16, C * C / 4);
  cvt_f32_to_f16<<<2048, 256, 0, stream>>>(W1, w116, H * C / 4);
  cvt_f32_to_f16<<<2048, 256, 0, stream>>>(W2, w216, C * H / 4);

  // zero q/k/v (padding rows must be 0 for attention masking)
  zero_kernel<<<2048, 256, 0, stream>>>((uint4*)(ws + o_q),
                                        (long)(3ull * BH * NPAD * 64 * 2 / 16));

  // LN1
  ln_kernel<<<M, 256, 0, stream>>>(x, g1, be1, h16);

  const dim3 gC((M + 127) / 128, C / 128);
  // QKV projections
  gemm_kernel<0><<<gC, 256, 0, stream>>>(h16, wq16, nullptr, nullptr, q16, M, C, C);
  gemm_kernel<1><<<gC, 256, 0, stream>>>(h16, wk16, nullptr, nullptr, k16, M, C, C);
  gemm_kernel<2><<<gC, 256, 0, stream>>>(h16, wv16, nullptr, nullptr, v16, M, C, C);

  // fused attention (BH heads, 37 query tiles / 8 waves -> grid.y = 5)
  attn_kernel<<<dim3(BH, 5), 256, 0, stream>>>(q16, k16, v16, o16);

  // output projection + bias + residual1 -> x1 (f32)
  gemm_kernel<3><<<gC, 256, 0, stream>>>(o16, wp16, bp, x, (void*)x1, M, C, C);

  // LN2 (reuse h16)
  ln_kernel<<<M, 256, 0, stream>>>(x1, g2, be2, h16);

  // MLP
  gemm_kernel<4><<<dim3((M + 127) / 128, H / 128), 256, 0, stream>>>(
      h16, w116, b1, nullptr, a16, M, H, C);
  gemm_kernel<5><<<gC, 256, 0, stream>>>(a16, w216, b2, x1, (void*)out, M, C, H);
}
